// Preprocess_Line2Window_7447473291370
// MI455X (gfx1250) — compile-verified, matrix operand order checked by reading, not run on hardware
//
#include <hip/hip_runtime.h>
#include <hip/hip_bf16.h>
#include <math.h>

// ---------------- problem constants (match reference) ----------------
#define B_    2
#define D_    128
#define HF_   120
#define WF_   120
#define NKP   1600            // HK*WK = 40*40
#define KWIN  15              // int(0.125 * 120)
#define K2W   (KWIN*KWIN)     // 225
#define TEMPR 10.0f
#define WINSZ 0.125f
#define CC    479.5f          // (960-1)/2 for all h/w = 960
#define EPSV  1e-6f

typedef __attribute__((ext_vector_type(16))) _Float16 v16h;
typedef __attribute__((ext_vector_type(8)))  float    v8f;
typedef __attribute__((ext_vector_type(8)))  _Float16 h8;

// ---------------- helpers ----------------
__device__ __forceinline__ void bilin(float xn, float yn,
    int& i00, int& i01, int& i10, int& i11,
    float& w00, float& w01, float& w10, float& w11) {
  float x = (xn + 1.0f) * ((WF_ - 1) * 0.5f);
  float y = (yn + 1.0f) * ((HF_ - 1) * 0.5f);
  float x0f = floorf(x), y0f = floorf(y);
  float wx = x - x0f, wy = y - y0f;
  int x0 = (int)x0f; x0 = x0 < 0 ? 0 : (x0 > WF_ - 1 ? WF_ - 1 : x0);
  int x1 = x0 + 1;   x1 = x1 > WF_ - 1 ? WF_ - 1 : x1;
  int y0 = (int)y0f; y0 = y0 < 0 ? 0 : (y0 > HF_ - 1 ? HF_ - 1 : y0);
  int y1 = y0 + 1;   y1 = y1 > HF_ - 1 ? HF_ - 1 : y1;
  i00 = y0 * WF_ + x0; i01 = y0 * WF_ + x1;
  i10 = y1 * WF_ + x0; i11 = y1 * WF_ + x1;
  w00 = (1.f - wx) * (1.f - wy); w01 = wx * (1.f - wy);
  w10 = (1.f - wx) * wy;         w11 = wx * wy;
}

__device__ __forceinline__ float blockReduceSum256(float v, float* red, int tid) {
  red[tid] = v; __syncthreads();
  for (int s = 128; s > 0; s >>= 1) { if (tid < s) red[tid] += red[tid + s]; __syncthreads(); }
  float r = red[0]; __syncthreads();
  return r;
}

__device__ __forceinline__ float blockReduceMax256(float v, float* red, int tid) {
  red[tid] = v; __syncthreads();
  for (int s = 128; s > 0; s >>= 1) { if (tid < s) red[tid] = fmaxf(red[tid], red[tid + s]); __syncthreads(); }
  float r = red[0]; __syncthreads();
  return r;
}

// ---------------- K1: sample descriptors at keypoints, l2norm, write f16 + pixel coords ----------------
__global__ void k_desc(const float* __restrict__ xf, const float* __restrict__ coord_n,
                       _Float16* __restrict__ feat16, float* __restrict__ coord_out) {
  int bn = blockIdx.x;               // b*NKP + n
  int b  = bn / NKP;
  int d  = threadIdx.x;              // 0..127 = channel
  float xn = coord_n[bn * 2 + 0];
  float yn = coord_n[bn * 2 + 1];
  int i00, i01, i10, i11; float w00, w01, w10, w11;
  bilin(xn, yn, i00, i01, i10, i11, w00, w01, w10, w11);
  const float* ch = xf + (((size_t)b * D_ + d) * (HF_ * WF_));
  float v = ch[i00] * w00 + ch[i01] * w01 + ch[i10] * w10 + ch[i11] * w11;
  __shared__ float red[128];
  red[d] = v * v; __syncthreads();
  for (int s = 64; s > 0; s >>= 1) { if (d < s) red[d] += red[d + s]; __syncthreads(); }
  float nrm = fmaxf(sqrtf(red[0]), 1e-12f);
  feat16[(size_t)bn * D_ + d] = (_Float16)(v / nrm);
  if (d < 2) coord_out[bn * 2 + d] = (d == 0 ? xn : yn) * CC + CC;
}

// ---------------- K2: t * l2norm(xf, channel) -> f16, channel-last (b,H,W,D) ----------------
__global__ void k_fmap(const float* __restrict__ xf, _Float16* __restrict__ out) {
  int p  = blockIdx.x;               // b*H*W + y*W + x
  int b  = p / (HF_ * WF_);
  int yx = p % (HF_ * WF_);
  int d  = threadIdx.x;
  float v = xf[((size_t)b * D_ + d) * (HF_ * WF_) + yx];
  __shared__ float red[128];
  red[d] = v * v; __syncthreads();
  for (int s = 64; s > 0; s >>= 1) { if (d < s) red[d] += red[d + s]; __syncthreads(); }
  float nrm = fmaxf(sqrtf(red[0]), 1e-12f);
  out[(size_t)p * D_ + d] = (_Float16)(TEMPR * v / nrm);
}

// ---------------- K3: sim = feat1 @ feat2^T via v_wmma_f32_16x16x32_f16 ----------------
// One wave -> a 16x64 output strip (4 adjacent 16x16 N-tiles): A fragment loaded once
// per k-step and reused across 4 WMMAs (4x less A traffic, 16 WMMA ops per wave).
// Fragment layout (CDNA5 16-bit A/B, K-major per lane): lane L handles row (L&15);
// half j of the v16h maps to K = (j&7) + 16*(j>>3) + 8*(L>>4).
__global__ void k_gemm(const _Float16* __restrict__ fa, const _Float16* __restrict__ fb,
                       float* __restrict__ sim) {
  const int TN = NKP / 16;           // 100 tiles per dim
  const int GN = TN / 4;             // 25 groups of 4 N-tiles
  int wave = (int)((blockIdx.x * blockDim.x + threadIdx.x) >> 5);
  int lane = threadIdx.x & 31;
  int total = B_ * TN * GN;          // 5000 waves
  if (wave >= total) return;         // uniform per-wave; grid sized exactly
  int b  = wave / (TN * GN);
  int t  = wave % (TN * GN);
  int tm = t / GN;                   // M tile
  int tg = t % GN;                   // group of 4 N tiles
  int r    = lane & 15;
  int hsel = lane >> 4;
  const _Float16* pa = fa + ((size_t)b * NKP + tm * 16 + r) * D_ + 8 * hsel;
  const _Float16* pb = fb + ((size_t)b * NKP + tg * 64 + r) * D_ + 8 * hsel;
  v8f acc0 = {}, acc1 = {}, acc2 = {}, acc3 = {};
#pragma unroll
  for (int k0 = 0; k0 < D_; k0 += 32) {
    union { v16h v; h8 h[2]; } A, B0, B1, B2, B3;
    A.h[0]  = *(const h8*)(pa + k0);
    A.h[1]  = *(const h8*)(pa + k0 + 16);
    B0.h[0] = *(const h8*)(pb + 0 * 16 * D_ + k0);
    B0.h[1] = *(const h8*)(pb + 0 * 16 * D_ + k0 + 16);
    B1.h[0] = *(const h8*)(pb + 1 * 16 * D_ + k0);
    B1.h[1] = *(const h8*)(pb + 1 * 16 * D_ + k0 + 16);
    B2.h[0] = *(const h8*)(pb + 2 * 16 * D_ + k0);
    B2.h[1] = *(const h8*)(pb + 2 * 16 * D_ + k0 + 16);
    B3.h[0] = *(const h8*)(pb + 3 * 16 * D_ + k0);
    B3.h[1] = *(const h8*)(pb + 3 * 16 * D_ + k0 + 16);
    acc0 = __builtin_amdgcn_wmma_f32_16x16x32_f16(false, A.v, false, B0.v, (short)0, acc0, false, false);
    acc1 = __builtin_amdgcn_wmma_f32_16x16x32_f16(false, A.v, false, B1.v, (short)0, acc1, false, false);
    acc2 = __builtin_amdgcn_wmma_f32_16x16x32_f16(false, A.v, false, B2.v, (short)0, acc2, false, false);
    acc3 = __builtin_amdgcn_wmma_f32_16x16x32_f16(false, A.v, false, B3.v, (short)0, acc3, false, false);
  }
  // D layout: VGPR rr -> M = rr + 8*(lane>>4); N = lane&15
  float* out = sim + ((size_t)b * NKP + (size_t)tm * 16) * NKP + (size_t)tg * 64;
  int mo = 8 * hsel, nc = lane & 15;
#pragma unroll
  for (int rr = 0; rr < 8; ++rr) {
    size_t rowoff = (size_t)(rr + mo) * NKP;
    out[rowoff +  0 + nc] = acc0[rr];
    out[rowoff + 16 + nc] = acc1[rr];
    out[rowoff + 32 + nc] = acc2[rr];
    out[rowoff + 48 + nc] = acc3[rr];
  }
}

// ---------------- K4: row softmax stats (p12): corloc + std for image1 keypoints ----------------
__global__ void k_rowstats(const float* __restrict__ sim, const float* __restrict__ cB_n,
                           float* __restrict__ corloc_out, float* __restrict__ std_out,
                           float* __restrict__ cn_ws) {
  int bn = blockIdx.x;               // b*NKP + n (row)
  int b  = bn / NKP;
  int tid = threadIdx.x;
  const float* row = sim + (size_t)bn * NKP;
  const float* cb  = cB_n + (size_t)b * NKP * 2;
  __shared__ float red[256];
  float mx = -1e30f;
  for (int m = tid; m < NKP; m += 256) mx = fmaxf(mx, row[m]);
  mx = blockReduceMax256(mx, red, tid);
  float se = 0, sx = 0, sy = 0, sxx = 0, syy = 0;
  for (int m = tid; m < NKP; m += 256) {
    float e  = expf(TEMPR * (row[m] - mx));
    float cx = cb[2 * m], cy = cb[2 * m + 1];
    se += e; sx += e * cx; sy += e * cy; sxx += e * cx * cx; syy += e * cy * cy;
  }
  se  = blockReduceSum256(se,  red, tid);
  sx  = blockReduceSum256(sx,  red, tid);
  sy  = blockReduceSum256(sy,  red, tid);
  sxx = blockReduceSum256(sxx, red, tid);
  syy = blockReduceSum256(syy, red, tid);
  if (tid == 0) {
    float cnx = sx / se, cny = sy / se;
    float vx = sxx / se - cnx * cnx, vy = syy / se - cny * cny;
    corloc_out[bn * 2 + 0] = cnx * CC + CC;
    corloc_out[bn * 2 + 1] = cny * CC + CC;
    std_out[bn] = sqrtf(fmaxf(vx, EPSV)) + sqrtf(fmaxf(vy, EPSV));
    cn_ws[bn * 2 + 0] = cnx;
    cn_ws[bn * 2 + 1] = cny;
  }
}

// ---------------- K5: column softmax stats (p21): corloc + std for image2 keypoints ----------------
__global__ void k_colstats(const float* __restrict__ sim, const float* __restrict__ cA_n,
                           float* __restrict__ corloc_out, float* __restrict__ std_out,
                           float* __restrict__ cn_ws) {
  int bm = blockIdx.x;               // b*NKP + m (column)
  int b  = bm / NKP;
  int m  = bm % NKP;
  int tid = threadIdx.x;
  const float* col = sim + (size_t)b * NKP * NKP + m;
  const float* ca  = cA_n + (size_t)b * NKP * 2;
  __shared__ float red[256];
  float mx = -1e30f;
  for (int n = tid; n < NKP; n += 256) mx = fmaxf(mx, col[(size_t)n * NKP]);
  mx = blockReduceMax256(mx, red, tid);
  float se = 0, sx = 0, sy = 0, sxx = 0, syy = 0;
  for (int n = tid; n < NKP; n += 256) {
    float e  = expf(TEMPR * (col[(size_t)n * NKP] - mx));
    float cx = ca[2 * n], cy = ca[2 * n + 1];
    se += e; sx += e * cx; sy += e * cy; sxx += e * cx * cx; syy += e * cy * cy;
  }
  se  = blockReduceSum256(se,  red, tid);
  sx  = blockReduceSum256(sx,  red, tid);
  sy  = blockReduceSum256(sy,  red, tid);
  sxx = blockReduceSum256(sxx, red, tid);
  syy = blockReduceSum256(syy, red, tid);
  if (tid == 0) {
    float cnx = sx / se, cny = sy / se;
    float vx = sxx / se - cnx * cnx, vy = syy / se - cny * cny;
    corloc_out[bm * 2 + 0] = cnx * CC + CC;
    corloc_out[bm * 2 + 1] = cny * CC + CC;
    std_out[bm] = sqrtf(fmaxf(vx, EPSV)) + sqrtf(fmaxf(vy, EPSV));
    cn_ws[bm * 2 + 0] = cnx;
    cn_ws[bm * 2 + 1] = cny;
  }
}

// ---------------- K6: window search (15x15 bilinear samples, softmax over window) ----------------
__global__ void k_window(const _Float16* __restrict__ feat,   // (b,N,128) descriptors (image A)
                         const _Float16* __restrict__ fmap,   // (b,H,W,128) t*l2norm (image B)
                         const float* __restrict__ center_n,  // (b,N,2) window centers
                         float* __restrict__ corloc_out, float* __restrict__ std_out) {
  int bn = blockIdx.x;
  int b  = bn / NKP;
  int tid = threadIdx.x;
  __shared__ __align__(16) _Float16 desc[D_];
  __shared__ float red[256];
  if (tid < D_) desc[tid] = feat[(size_t)bn * D_ + tid];
  float cx = center_n[bn * 2 + 0];
  float cy = center_n[bn * 2 + 1];
  __syncthreads();

  float simv = -1e30f, wcx = 0.f, wcy = 0.f;
  bool active = (tid < K2W);
  if (active) {
    int kx = tid % KWIN, ky = tid / KWIN;
    const float step = 2.f * WINSZ / (float)(KWIN - 1);
    wcx = cx + (-WINSZ + kx * step);
    wcy = cy + (-WINSZ + ky * step);
    int i00, i01, i10, i11; float w00, w01, w10, w11;
    bilin(wcx, wcy, i00, i01, i10, i11, w00, w01, w10, w11);
    const _Float16* base = fmap + (size_t)b * (HF_ * WF_) * D_;
    const _Float16* p00 = base + (size_t)i00 * D_;
    const _Float16* p01 = base + (size_t)i01 * D_;
    const _Float16* p10 = base + (size_t)i10 * D_;
    const _Float16* p11 = base + (size_t)i11 * D_;
    float acc = 0.f;
    for (int d0 = 0; d0 < D_; d0 += 8) {
      h8 a = *(const h8*)(p00 + d0);
      h8 bq = *(const h8*)(p01 + d0);
      h8 c = *(const h8*)(p10 + d0);
      h8 e = *(const h8*)(p11 + d0);
      h8 dd = *(const h8*)(desc + d0);
#pragma unroll
      for (int j = 0; j < 8; ++j) {
        float f = w00 * (float)a[j] + w01 * (float)bq[j] + w10 * (float)c[j] + w11 * (float)e[j];
        acc = fmaf(f, (float)dd[j], acc);
      }
    }
    simv = acc;
  }
  float mx = blockReduceMax256(simv, red, tid);
  float e  = active ? expf(simv - mx) : 0.f;
  float se  = blockReduceSum256(e,             red, tid);
  float sx  = blockReduceSum256(e * wcx,       red, tid);
  float sy  = blockReduceSum256(e * wcy,       red, tid);
  float sxx = blockReduceSum256(e * wcx * wcx, red, tid);
  float syy = blockReduceSum256(e * wcy * wcy, red, tid);
  if (tid == 0) {
    float cnx = sx / se, cny = sy / se;
    float vx = sxx / se - cnx * cnx, vy = syy / se - cny * cny;
    corloc_out[bn * 2 + 0] = cnx * CC + CC;
    corloc_out[bn * 2 + 1] = cny * CC + CC;
    std_out[bn] = sqrtf(fmaxf(vx, EPSV)) + sqrtf(fmaxf(vy, EPSV));
  }
}

// ---------------- host launch ----------------
extern "C" void kernel_launch(void* const* d_in, const int* in_sizes, int n_in,
                              void* d_out, int out_size, void* d_ws, size_t ws_size,
                              hipStream_t stream) {
  const float* xf1 = (const float*)d_in[0];
  const float* xf2 = (const float*)d_in[1];
  const float* c1n = (const float*)d_in[2];
  const float* c2n = (const float*)d_in[3];
  (void)in_sizes; (void)n_in; (void)out_size; (void)ws_size;

  float* out = (float*)d_out;
  // output layout (flat, return order)
  float* o_coord1   = out;                 // 6400
  float* o_coord2   = out + 6400;          // 6400
  float* o_f1g_cor  = out + 12800;         // 6400
  float* o_f2g_cor  = out + 19200;         // 6400
  float* o_f1w_cor  = out + 25600;         // 6400
  float* o_f2w_cor  = out + 32000;         // 6400
  float* o_f1g_std  = out + 38400;         // 3200
  float* o_f2g_std  = out + 41600;         // 3200
  float* o_f1w_std  = out + 44800;         // 3200
  float* o_f2w_std  = out + 48000;         // 3200

  // workspace layout (256-B aligned offsets)
  char* ws = (char*)d_ws;
  _Float16* feat1 = (_Float16*)(ws + 0);           //  819200 B
  _Float16* feat2 = (_Float16*)(ws + 819200);      //  819200 B
  _Float16* fmap1 = (_Float16*)(ws + 1638400);     // 7372800 B
  _Float16* fmap2 = (_Float16*)(ws + 9011200);     // 7372800 B
  float*    simm  = (float*)   (ws + 16384000);    // 20480000 B
  float*    cn1   = (float*)   (ws + 36864000);    //   25600 B (centers for image1 windows)
  float*    cn2   = (float*)   (ws + 36889600);    //   25600 B

  // 1) descriptors + pixel coords
  k_desc<<<B_ * NKP, 128, 0, stream>>>(xf1, c1n, feat1, o_coord1);
  k_desc<<<B_ * NKP, 128, 0, stream>>>(xf2, c2n, feat2, o_coord2);

  // 2) normalized temperature-scaled feature maps (channel-last f16)
  k_fmap<<<B_ * HF_ * WF_, 128, 0, stream>>>(xf1, fmap1);
  k_fmap<<<B_ * HF_ * WF_, 128, 0, stream>>>(xf2, fmap2);

  // 3) similarity GEMM: 2 * 100 * 25 wave-strips (16x64 each), 8 waves per block
  {
    int total_waves = B_ * (NKP / 16) * (NKP / 64);  // 5000
    int blocks = total_waves / 8;                    // 625
    k_gemm<<<blocks, 256, 0, stream>>>(feat1, feat2, simm);
  }

  // 4) grid softmax stats
  k_rowstats<<<B_ * NKP, 256, 0, stream>>>(simm, c2n, o_f1g_cor, o_f1g_std, cn1);
  k_colstats<<<B_ * NKP, 256, 0, stream>>>(simm, c1n, o_f2g_cor, o_f2g_std, cn2);

  // 5) window search (image1 descriptors vs featmap2, and vice versa)
  k_window<<<B_ * NKP, 256, 0, stream>>>(feat1, fmap2, cn1, o_f1w_cor, o_f1w_std);
  k_window<<<B_ * NKP, 256, 0, stream>>>(feat2, fmap1, cn2, o_f2w_cor, o_f2w_std);
}
